// DeepFloorFusedMixer_61289183314243
// MI455X (gfx1250) — matrix operand order repaired and checked by reading, not compiled, not run on hardware
//
#include <hip/hip_runtime.h>
#include <hip/hip_bf16.h>

typedef __attribute__((ext_vector_type(16))) __bf16 bf16x16;
typedef __attribute__((ext_vector_type(8)))  __bf16 bf16x8;
typedef __attribute__((ext_vector_type(8)))  float  f32x8;
typedef __attribute__((ext_vector_type(4)))  float  f32x4;
typedef int v4i_vs __attribute__((vector_size(16)));   // matches builtin param type

// ---------------- configuration ----------------
static constexpr int BM  = 128;    // block tile M
static constexpr int BN  = 128;    // block tile N
static constexpr int BK  = 32;     // block tile K (one WMMA K-step)
static constexpr int LDT = BK + 8; // LDS row stride (pad vs bank conflicts)
static constexpr float DECAYF = 0.95f;

#define AS_GLOBAL __attribute__((address_space(1)))
#define AS_LDS    __attribute__((address_space(3)))

#if __has_builtin(__builtin_amdgcn_global_load_async_to_lds_b128)
#define HAS_ASYNC 1
#else
#define HAS_ASYNC 0
#endif

template<int N>
__device__ __forceinline__ void wait_async() {
#if HAS_ASYNC
#if __has_builtin(__builtin_amdgcn_s_wait_asynccnt)
    __builtin_amdgcn_s_wait_asynccnt((unsigned short)N);
#else
    asm volatile("s_wait_asynccnt %0" :: "i"(N) : "memory");
#endif
#endif
}

// ---------------- helper kernels ----------------
__global__ __launch_bounds__(256)
void cvt_bf16_kernel(const float* __restrict__ in, __bf16* __restrict__ out, int n) {
    int i = (blockIdx.x * 256 + threadIdx.x) * 4;
    if (i + 3 >= n) return;
    f32x4 v = *(const f32x4*)(in + i);
    union { __bf16 b[4]; uint2 u; } u;
    u.b[0] = (__bf16)v.x; u.b[1] = (__bf16)v.y; u.b[2] = (__bf16)v.z; u.b[3] = (__bf16)v.w;
    *(uint2*)&out[i] = u.u;
}

// one block per row; D = 1024, 256 threads * 4 floats
__global__ __launch_bounds__(256)
void rmsnorm_kernel(const float* __restrict__ x, __bf16* __restrict__ out, int D) {
    long long row = blockIdx.x;
    const float* xr = x + row * (long long)D;
    int tid = threadIdx.x;
    f32x4 v = *(const f32x4*)(xr + tid * 4);
    float ss = v.x * v.x + v.y * v.y + v.z * v.z + v.w * v.w;
    #pragma unroll
    for (int o = 16; o > 0; o >>= 1) ss += __shfl_down(ss, o, 32);
    __shared__ float red[8];
    __shared__ float invs;
    if ((tid & 31) == 0) red[tid >> 5] = ss;
    __syncthreads();
    if (tid == 0) {
        float t = 0.f;
        #pragma unroll
        for (int i = 0; i < 8; ++i) t += red[i];
        invs = rsqrtf(t / (float)D + 1e-6f);
    }
    __syncthreads();
    float inv = invs;
    union { __bf16 b[4]; uint2 u; } u;
    u.b[0] = (__bf16)(v.x * inv); u.b[1] = (__bf16)(v.y * inv);
    u.b[2] = (__bf16)(v.z * inv); u.b[3] = (__bf16)(v.w * inv);
    *(uint2*)&out[row * (long long)D + tid * 4] = u.u;
}

// ---------------- bf16 WMMA GEMM ----------------
// 128 threads = 4 waves, 2x2 wave grid, each wave owns a 64x64 patch (16 WMMAs).
// TRANS == 0 : C[m,n] = sum_k Ag[m*lda+k] * Bg[n*ldb+k]   (NT form, async double-buffer)
// TRANS == 1 : C[m,n] = sum_k Ag[k*lda+m] * Bg[k*ldb+n]   (k^T v form, transpose staging)
// EPI: 0 = phi -> bf16 ; 1 = bf16 ; 2 = decay*extra + c*alpha -> f32 + transposed bf16
//      3 = c*alpha -> bf16 ; 4 = extra + c -> f32
template<int TRANS, int EPI>
__global__ __launch_bounds__(128)
void gemm_bf16_wmma(const __bf16* __restrict__ Ag, const __bf16* __restrict__ Bg,
                    int M, int N, int K, int lda, int ldb,
                    long long strideA, long long strideB,
                    const float* __restrict__ extra, long long strideE,
                    float* __restrict__ outF, long long strideF,
                    __bf16* __restrict__ outB, long long strideOB,
                    float alpha) {
    __shared__ __align__(16) __bf16 As[2][BM * LDT];
    __shared__ __align__(16) __bf16 Bs[2][BN * LDT];

    const int b = blockIdx.z;
    Ag += (long long)b * strideA;
    Bg += (long long)b * strideB;

    const int bm0 = blockIdx.y * BM;
    const int bn0 = blockIdx.x * BN;
    const int tid  = threadIdx.x;
    const int wid  = tid >> 5;
    const int lane = tid & 31;
    const int half = lane >> 4;        // 0: lanes 0-15, 1: lanes 16-31
    const int ln   = lane & 15;
    const int wrow0 = (wid >> 1) * 64; // 2x2 waves, 64x64 each
    const int wcol0 = (wid & 1) * 64;

    f32x8 acc[4][4];
    #pragma unroll
    for (int mt = 0; mt < 4; ++mt)
        #pragma unroll
        for (int nt = 0; nt < 4; ++nt)
            acc[mt][nt] = (f32x8)0.f;

    // ---- one WMMA compute step from LDS buffer `buf` ----
    auto compute = [&](int buf) {
        bf16x16 afrag[4];
        bf16x16 bfrag[4];
        #pragma unroll
        for (int mt = 0; mt < 4; ++mt) {
            int r  = wrow0 + mt * 16 + ln;
            int ko = half * 8;
            union { bf16x16 v; bf16x8 h[2]; } u;
            u.h[0] = *(const bf16x8*)&As[buf][r * LDT + ko];
            u.h[1] = *(const bf16x8*)&As[buf][r * LDT + ko + 16];
            afrag[mt] = u.v;
        }
        #pragma unroll
        for (int nt = 0; nt < 4; ++nt) {
            int r  = wcol0 + nt * 16 + ln;
            int ko = half * 8;
            union { bf16x16 v; bf16x8 h[2]; } u;
            u.h[0] = *(const bf16x8*)&Bs[buf][r * LDT + ko];
            u.h[1] = *(const bf16x8*)&Bs[buf][r * LDT + ko + 16];
            bfrag[nt] = u.v;
        }
        #pragma unroll
        for (int mt = 0; mt < 4; ++mt)
            #pragma unroll
            for (int nt = 0; nt < 4; ++nt)
                acc[mt][nt] = __builtin_amdgcn_wmma_f32_16x16x32_bf16(
                    false, afrag[mt], false, bfrag[nt],
                    (short)0, acc[mt][nt], false, false);
    };

    if (TRANS == 0) {
        // 512 16B chunks per tile; 128 threads -> 4 chunks each for A and B.
        auto stage = [&](int k0, int buf) {
            #pragma unroll
            for (int it = 0; it < 4; ++it) {
                int c   = tid + it * 128;
                int row = c >> 2;
                int kc  = (c & 3) * 8;
                const __bf16* pa = Ag + (long long)(bm0 + row) * lda + k0 + kc;
                const __bf16* pb = Bg + (long long)(bn0 + row) * ldb + k0 + kc;
#if HAS_ASYNC
                __builtin_amdgcn_global_load_async_to_lds_b128(
                    (AS_GLOBAL v4i_vs*)pa,
                    (AS_LDS v4i_vs*)&As[buf][row * LDT + kc], 0, 0);
                __builtin_amdgcn_global_load_async_to_lds_b128(
                    (AS_GLOBAL v4i_vs*)pb,
                    (AS_LDS v4i_vs*)&Bs[buf][row * LDT + kc], 0, 0);
#else
                *(uint4*)&As[buf][row * LDT + kc] = *(const uint4*)pa;
                *(uint4*)&Bs[buf][row * LDT + kc] = *(const uint4*)pb;
#endif
            }
        };
        const int kiters = K / BK;
        stage(0, 0);
        for (int i = 0; i < kiters; ++i) {
            int buf = i & 1;
            if (i + 1 < kiters) {
                stage((i + 1) * BK, buf ^ 1);
                wait_async<8>();   // 8 async ops/stage in flight for next stage
            } else {
                wait_async<0>();
            }
            __syncthreads();       // all waves' stage-i data visible in LDS
            compute(buf);
            __syncthreads();       // done reading buf before stage i+2 overwrites
        }
    } else {
        // transposed staging: global is [K][M] / [K][N]; scalar scatter into LDS
        for (int k0 = 0; k0 < K; k0 += BK) {
            #pragma unroll
            for (int it = 0; it < 4; ++it) {
                int c  = tid + it * 128;   // 512 chunks: 32 k-rows x 16 chunks of 8
                int kk = c >> 4;
                int m8 = (c & 15) * 8;
                bf16x8 va = *(const bf16x8*)(Ag + (long long)(k0 + kk) * lda + bm0 + m8);
                bf16x8 vb = *(const bf16x8*)(Bg + (long long)(k0 + kk) * ldb + bn0 + m8);
                #pragma unroll
                for (int i = 0; i < 8; ++i) {
                    As[0][(m8 + i) * LDT + kk] = va[i];
                    Bs[0][(m8 + i) * LDT + kk] = vb[i];
                }
            }
            __syncthreads();
            compute(0);
            __syncthreads();
        }
    }

    // ---- epilogue ----
    #pragma unroll
    for (int mt = 0; mt < 4; ++mt) {
        #pragma unroll
        for (int nt = 0; nt < 4; ++nt) {
            #pragma unroll
            for (int r = 0; r < 8; ++r) {
                int m = bm0 + wrow0 + mt * 16 + half * 8 + r;
                int n = bn0 + wcol0 + nt * 16 + ln;
                float c = acc[mt][nt][r];
                long long idx = (long long)m * N + n;
                if (EPI == 0) {
                    float p = c > 0.f ? c + 1.f : __expf(c);
                    outB[(long long)b * strideOB + idx] = (__bf16)p;
                } else if (EPI == 1) {
                    outB[(long long)b * strideOB + idx] = (__bf16)c;
                } else if (EPI == 2) {
                    float v = DECAYF * extra[(long long)b * strideE + idx] + c * alpha;
                    outF[(long long)b * strideF + idx] = v;
                    outB[(long long)b * strideOB + (long long)n * M + m] = (__bf16)v;
                } else if (EPI == 3) {
                    outB[(long long)b * strideOB + idx] = (__bf16)(c * alpha);
                } else {
                    outF[(long long)b * strideF + idx] =
                        extra[(long long)b * strideE + idx] + c;
                }
            }
        }
    }
}

// ---------------- launch ----------------
extern "C" void kernel_launch(void* const* d_in, const int* in_sizes, int n_in,
                              void* d_out, int out_size, void* d_ws, size_t ws_size,
                              hipStream_t stream) {
    (void)in_sizes; (void)n_in; (void)out_size; (void)ws_size;
    const int B = 4, S = 8192, D = 1024;
    const long long SD  = (long long)S * D;   // per-batch activation stride
    const long long DD  = (long long)D * D;
    const long long BSD = (long long)B * SD;

    const float* state  = (const float*)d_in[0];
    const float* acc_in = (const float*)d_in[1];
    const float* Wq     = (const float*)d_in[2];
    const float* Wk     = (const float*)d_in[3];
    const float* Wv     = (const float*)d_in[4];
    const float* Wo     = (const float*)d_in[5];

    float* out_f    = (float*)d_out;        // [B,S,D]
    float* acc_out  = out_f + BSD;          // [B,D,D]

    char* w = (char*)d_ws;
    __bf16* normed = (__bf16*)w;          w += BSD * 2;
    __bf16* qb     = (__bf16*)w;          w += BSD * 2;
    __bf16* kb     = (__bf16*)w;          w += BSD * 2;
    __bf16* vb     = (__bf16*)w;          w += BSD * 2;  // reused as `mixed`
    __bf16* accT   = (__bf16*)w;          w += (long long)B * DD * 2;
    __bf16* wqb    = (__bf16*)w;          w += DD * 2;
    __bf16* wkb    = (__bf16*)w;          w += DD * 2;
    __bf16* wvb    = (__bf16*)w;          w += DD * 2;
    __bf16* wob    = (__bf16*)w;

    // weights -> bf16
    int nW = (int)DD;
    int cb = nW / (256 * 4);
    cvt_bf16_kernel<<<cb, 256, 0, stream>>>(Wq, wqb, nW);
    cvt_bf16_kernel<<<cb, 256, 0, stream>>>(Wk, wkb, nW);
    cvt_bf16_kernel<<<cb, 256, 0, stream>>>(Wv, wvb, nW);
    cvt_bf16_kernel<<<cb, 256, 0, stream>>>(Wo, wob, nW);

    // rmsnorm -> bf16
    rmsnorm_kernel<<<B * S, 256, 0, stream>>>(state, normed, D);

    dim3 blk(128);

    // q/k/v projections: M=S (per batch), N=D, K=D ; B operand = weight [N,K] (NT)
    dim3 gproj(D / BN, S / BM, B);
    gemm_bf16_wmma<0, 0><<<gproj, blk, 0, stream>>>(normed, wqb, S, D, D, D, D,
        SD, 0, nullptr, 0, nullptr, 0, qb, SD, 0.f);
    gemm_bf16_wmma<0, 0><<<gproj, blk, 0, stream>>>(normed, wkb, S, D, D, D, D,
        SD, 0, nullptr, 0, nullptr, 0, kb, SD, 0.f);
    gemm_bf16_wmma<0, 1><<<gproj, blk, 0, stream>>>(normed, wvb, S, D, D, D, D,
        SD, 0, nullptr, 0, nullptr, 0, vb, SD, 0.f);

    // step = k^T v / S ; acc = decay*acc_in + step ; also write acc^T (bf16)
    dim3 gstep(D / BN, D / BM, B);
    gemm_bf16_wmma<1, 2><<<gstep, blk, 0, stream>>>(kb, vb, D, D, S, D, D,
        SD, SD, acc_in, DD, acc_out, DD, accT, DD, 1.f / (float)S);

    // mixed = q * acc / sqrt(D)   (acc^T is [e,d] row-major -> NT form), into vb
    gemm_bf16_wmma<0, 3><<<gproj, blk, 0, stream>>>(qb, accT, S, D, D, D, D,
        SD, DD, nullptr, 0, nullptr, 0, vb, SD, 0.03125f);

    // out = state + mixed * Wo^T
    gemm_bf16_wmma<0, 4><<<gproj, blk, 0, stream>>>(vb, wob, S, D, D, D, D,
        SD, 0, state, SD, out_f, SD, nullptr, 0, 0.f);
}
